// RelationNetworks_77687368450523
// MI455X (gfx1250) — matrix operand order repaired and checked by reading, not compiled, not run on hardware
//
#include <hip/hip_runtime.h>
#include <hip/hip_bf16.h>

typedef __attribute__((ext_vector_type(16))) _Float16 v16h;
typedef __attribute__((ext_vector_type(2)))  _Float16 v2h;
typedef __attribute__((ext_vector_type(8)))  float    v8f;

#define BN_EPS 1e-5f

__device__ __forceinline__ v8f wmma32f16(v16h a, v16h b, v8f c) {
  // D = A(16x32 f16) * B(32x16 f16) + C(16x16 f32)
  return __builtin_amdgcn_wmma_f32_16x16x32_f16(false, a, false, b, (short)0, c,
                                                false, false);
}

// Compiler-only ordering fence: same-wave LDS ops complete in order on CDNA5
// (DScnt, in-order LDS pipe), so store->load RAW within one wave's private
// h-tile needs no block barrier -- only that the compiler keeps program order.
__device__ __forceinline__ void lds_fence() { asm volatile("" ::: "memory"); }

// Swizzled f16-element index of A-matrix element (M,K) inside a 16x256 h-tile
// stored as contiguous WMMA A fragments (8 K-blocks of 32). From the CDNA5 ISA
// "16-bit A-Matrix 16x32" layout: lane holds row M=lane%16, dword i covers
// K = (i<4?0:16) + (lane<16?0:8) + (i%4)*2 (+half).
__device__ __forceinline__ int hswz(int M, int K) {
  int kt = K >> 5, kb = K & 31;
  int ln = (M & 15) + (((kb & 15) >= 8) ? 16 : 0);
  int dw = ((kb & 7) >> 1) + ((kb >= 16) ? 4 : 0);
  return ((kt * 256 + ln * 8 + dw) << 1) + (K & 1);
}

// ---------------------------------------------------------------------------
// Conv 3x3 stride-2 pad-1 + folded eval-mode BN + ReLU (tiny: ~2.5 GFLOP total)
// ---------------------------------------------------------------------------
__global__ void conv_bn_relu_k(const float* __restrict__ x, const float* __restrict__ w,
                               const float* __restrict__ g, const float* __restrict__ bb,
                               const float* __restrict__ bm, const float* __restrict__ bv,
                               float* __restrict__ y, int Cin, int Hin, int Hout)
{
  int i = blockIdx.x * 256 + threadIdx.x;
  int total = 128 * 24 * Hout * Hout;
  if (i >= total) return;
  int wo = i % Hout, t = i / Hout;
  int ho = t % Hout; t /= Hout;
  int co = t % 24;  int n = t / 24;
  float acc = 0.f;
  const float* wp = w + co * Cin * 9;
  const float* xp = x + (size_t)(n * Cin) * Hin * Hin;
  for (int ci = 0; ci < Cin; ++ci) {
    #pragma unroll
    for (int kh = 0; kh < 3; ++kh) {
      int hi = ho * 2 + kh - 1;
      if ((unsigned)hi >= (unsigned)Hin) continue;
      #pragma unroll
      for (int kw = 0; kw < 3; ++kw) {
        int wi = wo * 2 + kw - 1;
        if ((unsigned)wi >= (unsigned)Hin) continue;
        acc += xp[(ci * Hin + hi) * Hin + wi] * wp[ci * 9 + kh * 3 + kw];
      }
    }
  }
  float inv = g[co] * rsqrtf(bv[co] + BN_EPS);
  y[i] = fmaxf(acc * inv + (bb[co] - bm[co] * inv), 0.f);
}

// ---------------------------------------------------------------------------
// feats = [conv4, cx, cy];  U = feats @ gw1[0:26,:],  V = feats @ gw1[26:52,:]
// Collapses the 52-wide layer-1 GEMM over all 524288 pairs into two tiny
// per-object GEMMs: h1(pair a,c) = relu(U[c] + V[a] + b1).
// ---------------------------------------------------------------------------
__global__ void feats_uv_k(const float* __restrict__ x4, const float* __restrict__ gw1,
                           float* __restrict__ U, float* __restrict__ V)
{
  int i = blockIdx.x * 256 + threadIdx.x;        // 128*64*256
  int n = i & 255; int p = (i >> 8) & 63; int b = i >> 14;
  float au = 0.f, av = 0.f;
  #pragma unroll 4
  for (int c = 0; c < 24; ++c) {
    float f = x4[(b * 24 + c) * 64 + p];
    au += f * gw1[c * 256 + n];
    av += f * gw1[(26 + c) * 256 + n];
  }
  const float step = 8.0f / 7.0f;                // linspace(-4,4,8)
  float cx = -4.f + (float)(p & 7) * step;
  float cy = -4.f + (float)(p >> 3) * step;
  au += cx * gw1[24 * 256 + n] + cy * gw1[25 * 256 + n];
  av += cx * gw1[50 * 256 + n] + cy * gw1[51 * 256 + n];
  U[i] = au; V[i] = av;
}

// ---------------------------------------------------------------------------
// Swizzle a 256x256 f32 weight (K rows in, N cols out) into f16 WMMA B
// fragments: blocks [nt][kt], per-block lane-major 8 dwords (contiguous v16h
// per lane). B layout per ISA: lane gives N=lane%16 (+K-half), dwords walk K.
// ---------------------------------------------------------------------------
__global__ void wswz_k(const float* __restrict__ w, _Float16* __restrict__ ws)
{
  int i = blockIdx.x * 256 + threadIdx.x;        // 65536
  int K = i >> 8, N = i & 255;
  int nt = N >> 4, kt = K >> 5, kb = K & 31;
  int ln = (N & 15) + (((kb & 15) >= 8) ? 16 : 0);
  int dw = ((kb & 7) >> 1) + ((kb >= 16) ? 4 : 0);
  int idx = (((nt * 8 + kt) * 256) + ln * 8 + dw) * 2 + (kb & 1);
  ws[idx] = (_Float16)w[i];
}

// ---------------------------------------------------------------------------
// Fragment loads
// ---------------------------------------------------------------------------
__device__ __forceinline__ v16h ldA(const _Float16* ht, int lane, int kt) {
  return *(const v16h*)(ht + kt * 512 + lane * 16);
}
__device__ __forceinline__ v16h ldB(const _Float16* wm, int lane, int blk) {
  return *(const v16h*)(wm + blk * 512 + lane * 16);
}

// ---------------------------------------------------------------------------
// Fused pair-MLP layer: acc[nt] += A(ht) * B(wm) over K=256.
// Software-pipelined: depth-4 circular B prefetch + double-buffered A so DS /
// global loads overlap the matrix pipe (enables partial s_wait_dscnt).
// ---------------------------------------------------------------------------
__device__ __forceinline__ void mlp_layer_acc(const _Float16* ht, const _Float16* wm,
                                              int lane, v8f acc[16])
{
  v16h Bq[4];
  #pragma unroll
  for (int j = 0; j < 4; ++j)              // j<16 -> kt=0, nt=j
    Bq[j] = ldB(wm, lane, j << 3);
  v16h Acur = ldA(ht, lane, 0);
  v16h Anxt = ldA(ht, lane, 1);
  #pragma unroll
  for (int j = 0; j < 128; ++j) {
    const int kt = j >> 4, nt = j & 15;
    acc[nt] = wmma32f16(Acur, Bq[j & 3], acc[nt]);
    const int jn = j + 4;
    if (jn < 128) {
      const int ktn = jn >> 4, ntn = jn & 15;
      Bq[j & 3] = ldB(wm, lane, (ntn << 3) + ktn);
    }
    if (nt == 15) {
      Acur = Anxt;
      if (kt + 2 < 8) Anxt = ldA(ht, lane, kt + 2);
    }
  }
}

// relu(acc+bias) -> h-tile in swizzled A-fragment layout (f16 b16 stores).
// C layout per ISA: lane holds N=lane%16, rows M = v + (lane<16?0:8).
__device__ __forceinline__ void relu_store_ht(_Float16* ht, const float* __restrict__ bias,
                                              int lane, const v8f acc[16])
{
  int col = lane & 15, mb = (lane < 16) ? 0 : 8;
  #pragma unroll
  for (int nt = 0; nt < 16; ++nt) {
    float bs = bias[nt * 16 + col];
    #pragma unroll
    for (int v = 0; v < 8; ++v) {
      float x = fmaxf(acc[nt][v] + bs, 0.f);
      ht[hswz(mb + v, nt * 16 + col)] = (_Float16)x;
    }
  }
}

__device__ __forceinline__ void zero_acc(v8f acc[16]) {
  #pragma unroll
  for (int nt = 0; nt < 16; ++nt)
    #pragma unroll
    for (int v = 0; v < 8; ++v) acc[nt][v] = 0.f;
}

// ---------------------------------------------------------------------------
// Main fused pair-MLP: 206 GFLOP of fp16 WMMA, no pair tensor in HBM.
// LDS (320KB exactly): W2(128K) + W3(128K) + 8 per-wave 16x256 f16 h-tiles.
// W4 streamed from global/L2 (prefetched). 512 blocks x 8 waves; each wave
// owns 8 contiguous 16-pair tiles of one image; pair-sum folded row-wise per
// tile into 16 scalar registers per lane. Bulk copy / staging loops use small
// unroll factors to bound loads-in-flight and keep VGPRs < 256.
// ---------------------------------------------------------------------------
__launch_bounds__(256, 1)
__global__ void pair_mlp_k(const float* __restrict__ U, const float* __restrict__ V,
                           const float* __restrict__ gb1, const float* __restrict__ gb2,
                           const float* __restrict__ gb3, const float* __restrict__ gb4,
                           const _Float16* __restrict__ W2s, const _Float16* __restrict__ W3s,
                           const _Float16* __restrict__ W4s, float* __restrict__ gsum)
{
  extern __shared__ _Float16 lds[];
  _Float16* w2 = lds;                       // 65536 f16
  _Float16* w3 = lds + 65536;               // 65536 f16
  const int tid = threadIdx.x;
  const int wid = tid >> 5, lane = tid & 31;
  _Float16* ht = lds + 131072 + wid * 4096; // 16x256 f16 per wave

  // cooperative fill of W2/W3 into LDS (b128 copies; bounded unroll)
  {
    const uint4* s2 = (const uint4*)W2s;    // 8192 uint4 each
    const uint4* s3 = (const uint4*)W3s;
    uint4* d2 = (uint4*)w2;
    uint4* d3 = (uint4*)w3;
    #pragma unroll 4
    for (int i = tid; i < 8192; i += 256) { d2[i] = s2[i]; d3[i] = s3[i]; }
  }
  // prefetch layer-4 weights (128KB) toward the caches
  #pragma unroll 2
  for (int i = tid; i < 512; i += 256)
    __builtin_prefetch((const char*)W4s + i * 256, 0, 1);
  __syncthreads();                          // only true cross-wave dependency

  const int wave  = blockIdx.x * 8 + wid;   // 4096 waves total
  const int tile0 = wave * 8;               // 32768 tiles of 16 pairs
  const int b     = tile0 >> 8;             // 256 tiles per image
  const int col   = lane & 15;
  float gacc[16];
  #pragma unroll
  for (int nt = 0; nt < 16; ++nt) gacc[nt] = 0.f;

  for (int t = 0; t < 8; ++t) {
    const int prow0 = ((tile0 + t) & 255) * 16;
    // ---- layer 1: h1 = relu(U[c] + V[a] + b1), staged as A fragments.
    // Bounded unroll: caps loads-in-flight so VGPR pressure stays < 256.
    #pragma unroll 4
    for (int it = 0; it < 64; ++it) {
      int idx = it * 32 + lane;
      int r = idx >> 7, n = (idx & 127) * 2;
      int p = prow0 + r, a = p >> 6, c = p & 63;
      float2 uu = *(const float2*)(U + ((b * 64 + c) << 8) + n);
      float2 vv = *(const float2*)(V + ((b * 64 + a) << 8) + n);
      float2 bs = *(const float2*)(gb1 + n);
      v2h hv = { (_Float16)fmaxf(uu.x + vv.x + bs.x, 0.f),
                 (_Float16)fmaxf(uu.y + vv.y + bs.y, 0.f) };
      *(v2h*)(ht + hswz(r, n)) = hv;
    }
    lds_fence();
    // ---- layer 2 (weights in LDS)
    v8f acc[16];
    zero_acc(acc);
    mlp_layer_acc(ht, w2, lane, acc);
    lds_fence();
    relu_store_ht(ht, gb2, lane, acc);
    lds_fence();
    // ---- layer 3 (weights in LDS)
    zero_acc(acc);
    mlp_layer_acc(ht, w3, lane, acc);
    lds_fence();
    relu_store_ht(ht, gb3, lane, acc);
    lds_fence();
    // ---- layer 4 (weights streamed from global/L2), fold rows immediately
    zero_acc(acc);
    mlp_layer_acc(ht, W4s, lane, acc);
    #pragma unroll
    for (int nt = 0; nt < 16; ++nt) {
      float bs = gb4[nt * 16 + col];
      float s = 0.f;
      #pragma unroll
      for (int v = 0; v < 8; ++v) s += fmaxf(acc[nt][v] + bs, 0.f);
      gacc[nt] += s;
    }
    lds_fence();   // keep next tile's h1 stores after this tile's A reads
  }
  // flush per-wave pair-sum
  #pragma unroll
  for (int nt = 0; nt < 16; ++nt)
    atomicAdd(&gsum[b * 256 + nt * 16 + col], gacc[nt]);
}

// ---------------------------------------------------------------------------
// Tiny f-MLP head (35 MFLOP): plain fp32.
// ---------------------------------------------------------------------------
__global__ void fc_k(const float* __restrict__ x, const float* __restrict__ w,
                     const float* __restrict__ bias, float* __restrict__ y,
                     int Kd, int Nd, int doRelu)
{
  int i = blockIdx.x * 256 + threadIdx.x;
  if (i >= 128 * Nd) return;
  int n = i % Nd, r = i / Nd;
  float acc = bias[n];
  for (int k = 0; k < Kd; ++k) acc += x[r * Kd + k] * w[k * Nd + n];
  y[i] = doRelu ? fmaxf(acc, 0.f) : acc;
}

extern "C" void kernel_launch(void* const* d_in, const int* in_sizes, int n_in,
                              void* d_out, int out_size, void* d_ws, size_t ws_size,
                              hipStream_t stream)
{
  (void)in_sizes; (void)n_in; (void)out_size; (void)ws_size;
  const float* image = (const float*)d_in[0];
  const float *cw[4], *bg[4], *bb[4], *bm[4], *bv[4];
  for (int i = 0; i < 4; ++i) {
    cw[i] = (const float*)d_in[1 + i * 5];
    bg[i] = (const float*)d_in[2 + i * 5];
    bb[i] = (const float*)d_in[3 + i * 5];
    bm[i] = (const float*)d_in[4 + i * 5];
    bv[i] = (const float*)d_in[5 + i * 5];
  }
  const float* gw1 = (const float*)d_in[21];
  const float* gb1 = (const float*)d_in[22];
  const float* gw2 = (const float*)d_in[23];
  const float* gb2 = (const float*)d_in[24];
  const float* gw3 = (const float*)d_in[25];
  const float* gb3 = (const float*)d_in[26];
  const float* gw4 = (const float*)d_in[27];
  const float* gb4 = (const float*)d_in[28];
  const float* fw1 = (const float*)d_in[29];
  const float* fb1 = (const float*)d_in[30];
  const float* fw2 = (const float*)d_in[31];
  const float* fb2 = (const float*)d_in[32];
  const float* fw3 = (const float*)d_in[33];
  const float* fb3 = (const float*)d_in[34];

  float* ws = (float*)d_ws;
  float* x1 = ws;                          // 128*24*64*64 = 12,582,912
  float* x2 = x1 + 12582912;               // 128*24*32*32 =  3,145,728
  float* x3 = x2 + 3145728;                // 128*24*16*16 =    786,432
  float* x4 = x3 + 786432;                 // 128*24*8*8   =    196,608
  float* U  = x1;                          // alias: x1 dead after conv2
  float* V  = x1 + 2097152;                // 128*64*256 each
  float* gsum = x4 + 196608;               // 128*256
  float* y1 = gsum + 32768;
  float* y2 = y1 + 32768;
  _Float16* W2s = (_Float16*)(y2 + 32768); // 65536 f16 each, swizzled
  _Float16* W3s = W2s + 65536;
  _Float16* W4s = W3s + 65536;

  // conv tower
  conv_bn_relu_k<<<49152, 256, 0, stream>>>(image, cw[0], bg[0], bb[0], bm[0], bv[0], x1, 3, 128, 64);
  conv_bn_relu_k<<<12288, 256, 0, stream>>>(x1, cw[1], bg[1], bb[1], bm[1], bv[1], x2, 24, 64, 32);
  conv_bn_relu_k<<<3072, 256, 0, stream>>>(x2, cw[2], bg[2], bb[2], bm[2], bv[2], x3, 24, 32, 16);
  conv_bn_relu_k<<<768, 256, 0, stream>>>(x3, cw[3], bg[3], bb[3], bm[3], bv[3], x4, 24, 16, 8);

  // per-object layer-1 factorization
  feats_uv_k<<<8192, 256, 0, stream>>>(x4, gw1, U, V);

  // weight swizzle into WMMA B-fragment layout (f16)
  wswz_k<<<256, 256, 0, stream>>>(gw2, W2s);
  wswz_k<<<256, 256, 0, stream>>>(gw3, W3s);
  wswz_k<<<256, 256, 0, stream>>>(gw4, W4s);

  hipMemsetAsync(gsum, 0, 32768 * sizeof(float), stream);

  // fused WMMA pair-MLP with pair-sum reduction (320KB dynamic LDS)
  pair_mlp_k<<<512, 256, 327680, stream>>>(U, V, gb1, gb2, gb3, gb4,
                                           W2s, W3s, W4s, gsum);

  // f head
  fc_k<<<128, 256, 0, stream>>>(gsum, fw1, fb1, y1, 256, 256, 1);
  fc_k<<<128, 256, 0, stream>>>(y1, fw2, fb2, y2, 256, 256, 1);
  fc_k<<<15, 256, 0, stream>>>(y2, fw3, fb3, (float*)d_out, 256, 29, 0);
}
